// DifferentiableAstar_54769422959089
// MI455X (gfx1250) — matrix operand order repaired and checked by reading, not compiled, not run on hardware
//
#include <hip/hip_runtime.h>
#include <stdint.h>

// ---------------------------------------------------------------------------
// Differentiable A* on MI455X (gfx1250).
//  * grid = 16 workgroups (one per batch), 256 threads = 8 wave32 each
//  * all per-batch state lives in LDS (128KB of the 320KB/WGP)
//  * inputs staged global->LDS via the Tensor Data Mover (TENSORcnt)
//  * goal-location einsum ('kij,bij->bk') done with v_wmma_f32_16x16x32_f16
//  * global all_solved consensus via a generation barrier in d_ws
// ---------------------------------------------------------------------------

#define B_    16
#define SZ    64
#define HW    4096          // 64*64
#define STEPS 409           // int(0.1 * 64 * 64)
#define NT    256

typedef __attribute__((ext_vector_type(16))) _Float16     v16h;
typedef __attribute__((ext_vector_type(8)))  float        v8f;
typedef __attribute__((ext_vector_type(4)))  unsigned int u32x4;
typedef __attribute__((ext_vector_type(8)))  int          i32x8;
typedef __attribute__((ext_vector_type(4)))  int          i32x4;

struct RedBlock {
    float redv[8];
    int   redi[8];
    float gr, gc;
    int   goal_idx;
    int   sel;
    int   flag;
};

struct GSync {             // lives at d_ws[0..7], zeroed by init kernel
    int arrive;            // monotonic arrival counter
    int release;           // monotonic release generation
    int solved[2];         // per-parity solved-batch counters
    int allflag[2];        // per-parity all_solved result
};

// Generic (flat) pointers into LDS carry the wave-relative LDS byte offset in
// their low 32 bits (CDNA5 ISA 10.2).
__device__ __forceinline__ uint32_t lds_raw_addr(const void* p) {
    return (uint32_t)(uintptr_t)p;
}

// Issue one TDM DMA: contiguous run of `nelem` f32 from global -> LDS.
// D# built per CDNA5 ISA 8.3/8.4 (group0: count/lds_addr/global_addr/type,
// group1: data_size=4B, tensor_dim0=tile_dim0=nelem, 1-D tile).
// This toolchain's builtin takes 6 args (groups 0..3 + extra group + cpol).
__device__ __forceinline__ void tdm_load_1d_f32(uint32_t lds_addr,
                                                const float* gptr,
                                                uint32_t nelem) {
    uint64_t ga = (uint64_t)(uintptr_t)gptr;
    u32x4 g0;
    g0[0] = 1u;                                      // count=1, user descriptor
    g0[1] = lds_addr;                                // D#.lds_addr
    g0[2] = (uint32_t)(ga & 0xffffffffull);          // global_addr[31:0]
    g0[3] = (uint32_t)((ga >> 32) & 0x01ffffffull)   // global_addr[56:32]
          | (2u << 30);                              // type=2 ("image")
    i32x8 g1;
    g1[0] = (int)(2u << 16);                         // data_size=2 (4B), wg_mask=0
    g1[1] = (int)((nelem & 0xffffu) << 16);          // tensor_dim0[15:0]
    g1[2] = (int)((nelem >> 16) & 0xffffu);          // tensor_dim0[31:16], dim1=0
    g1[3] = (int)((nelem & 0xffffu) << 16);          // tile_dim0 (<=65535)
    g1[4] = 0;                                       // tile_dim1=0, tile_dim2=0
    g1[5] = (int)nelem;                              // tensor_dim0_stride[31:0]
    g1[6] = 0;
    g1[7] = 0;
    i32x4 z4; z4[0] = 0; z4[1] = 0; z4[2] = 0; z4[3] = 0;   // groups 2/3 unused
    i32x8 z8; z8[0] = 0; z8[1] = 0; z8[2] = 0; z8[3] = 0;
    z8[4] = 0; z8[5] = 0; z8[6] = 0; z8[7] = 0;
    __builtin_amdgcn_tensor_load_to_lds(g0, g1, z4, z4, z8, 0);
}

// goal_loc einsum via WMMA (exact: all operands are small integers / {0,1}).
//   D[m,col] = sum_row A[m,row] * goal[row,col], A[0,row]=row, A[1,row]=1
//   => gr = sum_col D[0,col],  gc = sum_col col * D[1,col]
// K covers 64 rows in two 16x16x32 accumulations; N covers 64 cols in 4 tiles.
__device__ __forceinline__ void goal_loc_wmma(const float* __restrict__ gl,
                                              float* r_out, float* c_out) {
    const int lane = (int)(threadIdx.x & 31);
    const int m    = lane & 15;    // A row (lanes hold M=0..15) and B column
    const int grp  = lane >> 4;    // K-group select per ISA VGPR layout
    v8f zero = {0.f, 0.f, 0.f, 0.f, 0.f, 0.f, 0.f, 0.f};
    v8f acc[4] = {zero, zero, zero, zero};

#pragma unroll
    for (int rk = 0; rk < 2; ++rk) {
        // A fragment, 16x32: lanes 0-15 K=0..15, lanes 16-31 K=16..31
        // (contiguous order, matching B).
        v16h a;
#pragma unroll
        for (int h = 0; h < 16; ++h) {
            const int K = 16 * grp + h;
            float av = 0.f;
            if (m == 0)      av = (float)(32 * rk + K);  // row index weights
            else if (m == 1) av = 1.f;                   // column-mass weights
            a[h] = (_Float16)av;
        }
#pragma unroll
        for (int ct = 0; ct < 4; ++ct) {
            // B fragment, 32x16: lanes 0-15 K=0..15, lanes 16-31 K=16..31.
            v16h bm;
#pragma unroll
            for (int h = 0; h < 16; ++h) {
                const int K = 16 * grp + h;
                bm[h] = (_Float16)gl[(32 * rk + K) * SZ + ct * 16 + m];
            }
            acc[ct] = __builtin_amdgcn_wmma_f32_16x16x32_f16(
                false, a, false, bm, (short)0, acc[ct], false, false);
        }
    }
    // D layout: lanes 0-15 -> VGPR v holds D[v, lane]; lanes 16-31 hold rows
    // 8..15 which are zero rows of A, so they contribute nothing.
    float pr = acc[0][0] + acc[1][0] + acc[2][0] + acc[3][0];
    float pc = 0.f;
#pragma unroll
    for (int ct = 0; ct < 4; ++ct) pc += (float)(ct * 16 + m) * acc[ct][1];
#pragma unroll
    for (int off = 16; off > 0; off >>= 1) {
        pr += __shfl_xor(pr, off, 32);
        pc += __shfl_xor(pc, off, 32);
    }
    *r_out = pr;
    *c_out = pc;
}

__global__ void astar_sync_init(int* ws) {
    if (threadIdx.x < 8) ws[threadIdx.x] = 0;
}

__global__ __launch_bounds__(NT, 1)
void astar_kernel(const float* __restrict__ cost_g,
                  const float* __restrict__ start_g,
                  const float* __restrict__ goal_g,
                  const float* __restrict__ obst_g,
                  float* __restrict__ out,
                  int* __restrict__ wsync) {
    extern __shared__ float smem[];
    float* cost_l = smem;                 // cost map
    float* obst_l = cost_l + HW;          // obstacle map
    float* g_l    = obst_l + HW;          // g values
    float* sm_l   = g_l + HW;             // open set (also path map at end)
    float* hist_l = sm_l + HW;            // closed set
    float* hcost  = hist_l + HW;          // goal map during setup, then h+cost
    int*   par_l  = (int*)(hcost + HW);   // parent indices
    RedBlock* rb  = (RedBlock*)(par_l + HW);

    const int tid = (int)threadIdx.x;
    const int b   = (int)blockIdx.x;

    // ---- Stage the four input maps into LDS with the Tensor Data Mover ----
    if (tid < 32) {
        tdm_load_1d_f32(lds_raw_addr(cost_l), cost_g  + b * HW, HW);
        tdm_load_1d_f32(lds_raw_addr(sm_l),   start_g + b * HW, HW); // sm0=start
        tdm_load_1d_f32(lds_raw_addr(hcost),  goal_g  + b * HW, HW); // goal
        tdm_load_1d_f32(lds_raw_addr(obst_l), obst_g  + b * HW, HW);
        __builtin_amdgcn_s_wait_tensorcnt(0);
    }
    __syncthreads();

    // goal index (argmax of a one-hot map)
    for (int k = tid; k < HW; k += NT)
        if (hcost[k] > 0.5f) rb->goal_idx = k;
    __syncthreads();
    const int goal_idx = rb->goal_idx;

    // goal_loc einsum on the matrix core (wave 0, EXEC all-ones inside branch)
    if (tid < 32) {
        float r, c;
        goal_loc_wmma(hcost, &r, &c);
        if (tid == 0) { rb->gr = r; rb->gc = c; }
    }
    __syncthreads();
    const float gr = rb->gr, gc = rb->gc;

    // heuristic + state init; hcost[] = (manh - min + 0.001*euc) + cost
    for (int k = tid; k < HW; k += NT) {
        float fi = (float)(k >> 6);
        float fj = (float)(k & (SZ - 1));
        float di = fabsf(fi - gr), dj = fabsf(fj - gc);
        float hr = (di + dj - fminf(di, dj)) + 0.001f * sqrtf(di * di + dj * dj);
        hcost[k]  = hr + cost_l[k];
        g_l[k]    = 0.f;
        hist_l[k] = 0.f;
        par_l[k]  = goal_idx;
        // sm_l[k] already holds the start map (TDM)
    }
    __syncthreads();

    int t_stop = STEPS - 1;
    for (int t = 0; t < STEPS; ++t) {
        // argmax of exp(-(0.5g+0.5h)/8)*sm, first-index tie-break
        float bv = -1.0f; int bi = HW;
        for (int k = tid; k < HW; k += NT) {
            float f = 0.5f * g_l[k] + 0.5f * hcost[k];
            float e = expf(-f * 0.125f) * sm_l[k];
            if (e > bv) { bv = e; bi = k; }
        }
#pragma unroll
        for (int off = 16; off > 0; off >>= 1) {
            float ov = __shfl_xor(bv, off, 32);
            int   oi = __shfl_xor(bi, off, 32);
            if (ov > bv || (ov == bv && oi < bi)) { bv = ov; bi = oi; }
        }
        if ((tid & 31) == 0) { rb->redv[tid >> 5] = bv; rb->redi[tid >> 5] = bi; }
        __syncthreads();

        if (tid == 0) {
            float v = rb->redv[0]; int ix = rb->redi[0];
            for (int w = 1; w < NT / 32; ++w)
                if (rb->redv[w] > v || (rb->redv[w] == v && rb->redi[w] < ix)) {
                    v = rb->redv[w]; ix = rb->redi[w];
                }
            const int  s        = ix;
            const bool solved_b = (s == goal_idx);

            // sel is one-hot: apply the whole step as an O(8) update
            hist_l[s] = fminf(hist_l[s] + 1.f, 1.f);
            if (!solved_b) sm_l[s] = 0.f;
            const float gs2 = g_l[s] + cost_l[s];
            const int si = s >> 6, sj = s & (SZ - 1);
#pragma unroll
            for (int dy = -1; dy <= 1; ++dy)
#pragma unroll
                for (int dx = -1; dx <= 1; ++dx) {
                    if (dx == 0 && dy == 0) continue;
                    int ni = si + dy, nj = sj + dx;
                    if ((unsigned)ni < SZ && (unsigned)nj < SZ) {
                        int n = ni * SZ + nj;
                        float idxv = ((1.f - sm_l[n]) * (1.f - hist_l[n]) +
                                      sm_l[n] * ((g_l[n] > gs2) ? 1.f : 0.f)) *
                                     obst_l[n];
                        if (idxv > 0.f) { g_l[n] = gs2; sm_l[n] = 1.f; par_l[n] = s; }
                    }
                }

            // device-wide all_solved consensus (16 co-resident workgroups)
            GSync* gs = (GSync*)wsync;
            const int p = t & 1;
            __hip_atomic_fetch_add(&gs->solved[p], solved_b ? 1 : 0,
                                   __ATOMIC_RELAXED, __HIP_MEMORY_SCOPE_AGENT);
            int a = __hip_atomic_fetch_add(&gs->arrive, 1, __ATOMIC_ACQ_REL,
                                           __HIP_MEMORY_SCOPE_AGENT) + 1;
            if (a == B_ * (t + 1)) {
                int sc = __hip_atomic_load(&gs->solved[p], __ATOMIC_RELAXED,
                                           __HIP_MEMORY_SCOPE_AGENT);
                __hip_atomic_store(&gs->solved[p ^ 1], 0, __ATOMIC_RELAXED,
                                   __HIP_MEMORY_SCOPE_AGENT);
                __hip_atomic_store(&gs->allflag[p], (sc == B_) ? 1 : 0,
                                   __ATOMIC_RELAXED, __HIP_MEMORY_SCOPE_AGENT);
                __hip_atomic_store(&gs->release, t + 1, __ATOMIC_RELEASE,
                                   __HIP_MEMORY_SCOPE_AGENT);
            } else {
                while (__hip_atomic_load(&gs->release, __ATOMIC_ACQUIRE,
                                         __HIP_MEMORY_SCOPE_AGENT) < t + 1)
                    __builtin_amdgcn_s_sleep(8);
            }
            rb->flag = __hip_atomic_load(&gs->allflag[p], __ATOMIC_RELAXED,
                                         __HIP_MEMORY_SCOPE_AGENT);
        }
        __syncthreads();
        if (rb->flag) { t_stop = t; break; }   // state of step t already applied
    }

    // ---- outputs: hist then path (float 0/1 maps) ----
    for (int k = tid; k < HW; k += NT) out[b * HW + k] = hist_l[k];

    for (int k = tid; k < HW; k += NT) sm_l[k] = 0.f;  // reuse sm as path map
    __syncthreads();
    if (tid == 0) {
        sm_l[goal_idx] = 1.f;
        int loc = par_l[goal_idx];
        for (int i = 0; i < STEPS; ++i) {
            if (i < t_stop) { sm_l[loc] = 1.f; loc = par_l[loc]; }
        }
    }
    __syncthreads();
    for (int k = tid; k < HW; k += NT) out[B_ * HW + b * HW + k] = sm_l[k];
}

extern "C" void kernel_launch(void* const* d_in, const int* in_sizes, int n_in,
                              void* d_out, int out_size, void* d_ws, size_t ws_size,
                              hipStream_t stream) {
    (void)in_sizes; (void)n_in; (void)out_size; (void)ws_size;
    const float* cost_g  = (const float*)d_in[0];
    const float* start_g = (const float*)d_in[1];
    const float* goal_g  = (const float*)d_in[2];
    const float* obst_g  = (const float*)d_in[3];
    float* out   = (float*)d_out;
    int*   wsync = (int*)d_ws;

    const size_t shbytes = (size_t)(7 * HW) * sizeof(float) +
                           (size_t)HW * sizeof(int) + sizeof(RedBlock) + 64;
    (void)hipFuncSetAttribute((const void*)astar_kernel,
                              hipFuncAttributeMaxDynamicSharedMemorySize,
                              (int)shbytes);

    astar_sync_init<<<1, 64, 0, stream>>>(wsync);
    astar_kernel<<<B_, NT, shbytes, stream>>>(cost_g, start_g, goal_g, obst_g,
                                              out, wsync);
}
